// PairwiseRelationModule_42906723287623
// MI455X (gfx1250) — compile-verified, matrix operand order checked by reading, not run on hardware
//
#include <hip/hip_runtime.h>

typedef unsigned short u16;
typedef __attribute__((ext_vector_type(16))) __bf16 v16bf;
typedef __attribute__((ext_vector_type(8)))  float  v8f;
typedef __attribute__((ext_vector_type(4)))  unsigned int v4u;
typedef __attribute__((ext_vector_type(8)))  int v8i;
typedef __attribute__((ext_vector_type(4)))  int v4i;

#define B_   64
#define O_   36
#define D_   512
#define P_   (O_ * O_)     // 1296
#define ROWS1 (B_ * O_)    // 2304
#define BN_EPS 1e-5f

// ---------- helpers ----------

__device__ __forceinline__ u16 f2bf(float f) {
  unsigned int u = __float_as_uint(f);
  u += 0x7fffu + ((u >> 16) & 1u);   // round-to-nearest-even
  return (u16)(u >> 16);
}

// Element index inside a WMMA bf16 fragment-packed matrix.
// A-matrix: (row r, k) of an MxK matrix; B-matrix: call with r = n (col), k.
// Layout: [tile16][tileK32][lane 0..31][16 halves], 32B per lane per fragment.
// Per ISA 16-bit A layout: lanes 0-15 hold K 0-7,16-23; lanes 16-31 hold K 8-15,24-31.
__device__ __forceinline__ size_t frag_idx(int r, int k, int K) {
  int tM = r >> 4, tK = k >> 5, kl = k & 31;
  int lane = (r & 15) | (((kl >> 3) & 1) << 4);
  int j = (kl & 7) | ((kl & 16) >> 1);
  return ((((size_t)tM * (K >> 5) + tK) * 32 + lane) << 4) + (size_t)j;
}

__device__ __forceinline__ v8f wmma_bf16(v16bf a, v16bf b, v8f c) {
  return __builtin_amdgcn_wmma_f32_16x16x32_bf16(false, a, false, b, (short)0, c,
                                                 false, false);
}

// ---------- packing kernels (fp32 -> bf16 fragment layouts) ----------

__global__ void k_pack_objs(const float* __restrict__ objs, u16* __restrict__ dst) {
  int idx = blockIdx.x * blockDim.x + threadIdx.x;   // over 2304*512
  if (idx >= ROWS1 * D_) return;
  int r = idx >> 9, k = idx & 511;
  dst[frag_idx(r, k, D_)] = f2bf(objs[idx]);
}

// Layer-1 B matrix: (k=d, n), n<512 -> W1[n][d] (U half), n>=512 -> W1[n-512][512+d] (V half)
__global__ void k_pack_w1(const float* __restrict__ W1, u16* __restrict__ dst) {
  int idx = blockIdx.x * blockDim.x + threadIdx.x;   // over 1024*512
  if (idx >= 1024 * 512) return;
  int n = idx >> 9, k = idx & 511;
  float v = (n < 512) ? W1[(size_t)n * 1024 + k]
                      : W1[(size_t)(n - 512) * 1024 + 512 + k];
  dst[frag_idx(n, k, 512)] = f2bf(v);
}

// Layer-2 B matrix: (k=g1, n=g2) = W2[n][k]
__global__ void k_pack_w2(const float* __restrict__ W2, u16* __restrict__ dst) {
  int idx = blockIdx.x * blockDim.x + threadIdx.x;   // over 512*512 (idx = n*512+k)
  if (idx >= 512 * 512) return;
  int n = idx >> 9, k = idx & 511;
  dst[frag_idx(n, k, 512)] = f2bf(W2[idx]);
}

// ---------- GEMM1: UV[2304 x 1024] = objs(2304x512) @ [W1a^T | W1b^T] ----------
// WG = 256 thr (8 waves); per WG 64 rows x 128 cols; wave = 16 rows x 64 cols.
// K loop is double-buffered in registers so WMMAs don't wait per fragment.

__global__ void k_gemm1(const u16* __restrict__ Af, const u16* __restrict__ Bf,
                        float* __restrict__ UV) {
  const int w = threadIdx.x >> 5, lane = threadIdx.x & 31;
  const int mi = w >> 1, ng = w & 1;
  const int tM = blockIdx.x * 4 + mi;        // 0..143
  const int tN0 = blockIdx.y * 8 + ng * 4;   // N-tile base (of 64)

  v8f z = {0.f, 0.f, 0.f, 0.f, 0.f, 0.f, 0.f, 0.f};
  v8f acc[4];
#pragma unroll
  for (int ni = 0; ni < 4; ++ni) acc[ni] = z;

  auto LOAD = [&](v16bf& a, v16bf (&bv)[4], int kk) {
    a = *reinterpret_cast<const v16bf*>(Af + (((size_t)tM * 16 + kk) * 32 + lane) * 16);
#pragma unroll
    for (int ni = 0; ni < 4; ++ni)
      bv[ni] = *reinterpret_cast<const v16bf*>(
          Bf + (((size_t)(tN0 + ni) * 16 + kk) * 32 + lane) * 16);
  };
  auto MMA = [&](v16bf& a, v16bf (&bv)[4]) {
#pragma unroll
    for (int ni = 0; ni < 4; ++ni) acc[ni] = wmma_bf16(a, bv[ni], acc[ni]);
  };

  v16bf a0, a1, b0[4], b1[4];
  LOAD(a0, b0, 0);
  for (int kk = 0; kk < 14; kk += 2) {
    LOAD(a1, b1, kk + 1);
    MMA(a0, b0);
    LOAD(a0, b0, kk + 2);
    MMA(a1, b1);
  }
  LOAD(a1, b1, 15);
  MMA(a0, b0);
  MMA(a1, b1);

  const int row0 = tM * 16 + (lane >> 4) * 8;
  const int col0 = tN0 * 16 + (lane & 15);
#pragma unroll
  for (int ni = 0; ni < 4; ++ni)
#pragma unroll
    for (int i = 0; i < 8; ++i)
      UV[(size_t)(row0 + i) * 1024 + col0 + ni * 16] = acc[ni][i];
}

// ---------- GEMM2 (fused): h = relu(U[m]+V[k]+b1); x = relu(h @ W2^T + b2) ----------
// grid (21, 64): blockIdx.y = batch, blockIdx.x = 64-pair tile. 256 thr = 8 waves.
// LDS: h fragment tile 64x512 bf16 (64KB) + U 36x512 fp32 (72KB), dynamic.
// Wave owns 64 rows x 64 cols = 16 accumulator tiles; K double-buffered.

__global__ void k_gemm2(const float* __restrict__ UV, const float* __restrict__ b1,
                        const u16* __restrict__ W2f, const float* __restrict__ b2,
                        float* __restrict__ xp) {
  extern __shared__ char smem[];
  u16*   hf = (u16*)smem;                 // 65536 B
  float* Ul = (float*)(smem + 65536);     // 73728 B

  const int b   = blockIdx.y;
  const int p0  = blockIdx.x * 64;
  const int tid = threadIdx.x;

  // Stage U[b, 0:36, 0:512] (row stride 1024 in UV) into LDS via TDM.
#if __has_builtin(__builtin_amdgcn_tensor_load_to_lds)
  if (tid < 32) {   // one wave issues the DMA
    unsigned long long ga = (unsigned long long)(size_t)(const void*)(UV + (size_t)b * O_ * 1024);
    unsigned ldsa = (unsigned)(unsigned long long)(size_t)(void*)Ul;
    v4u g0; v8i g1; v4i g2, g3; v8i g4;
    g0[0] = 1u;                                  // count=1, user descriptor
    g0[1] = ldsa;                                // lds_addr
    g0[2] = (unsigned)(ga & 0xffffffffull);      // global_addr[31:0]
    g0[3] = (unsigned)((ga >> 32) & 0x1ffffffull) | 0x80000000u;  // addr[56:32], type=2
    g1[0] = 0x00020000;                          // data_size = 4B
    g1[1] = (int)(512u << 16);                   // tensor_dim0 = 512
    g1[2] = (int)((unsigned)O_ << 16);           // tensor_dim1 = 36
    g1[3] = (int)(512u << 16);                   // tile_dim0 = 512
    g1[4] = O_;                                  // tile_dim1 = 36
    g1[5] = 1024;                                // tensor_dim0_stride = 1024
    g1[6] = 0; g1[7] = 0;
    g2[0] = g2[1] = g2[2] = g2[3] = 0;
    g3[0] = g3[1] = g3[2] = g3[3] = 0;
    g4[0] = g4[1] = g4[2] = g4[3] = 0;
    g4[4] = g4[5] = g4[6] = g4[7] = 0;
    __builtin_amdgcn_tensor_load_to_lds(g0, g1, g2, g3, g4, 0);
    __builtin_amdgcn_s_wait_tensorcnt(0);
  }
#else
  for (int t = tid; t < O_ * 512; t += blockDim.x)
    Ul[t] = UV[(size_t)(b * O_ + (t >> 9)) * 1024 + (t & 511)];
#endif
  __syncthreads();

  // Build 64x512 bf16 h tile in fragment order.
  for (int t = tid; t < 64 * 512; t += 256) {
    int r = t >> 9, g = t & 511;
    int p = p0 + r;
    float hv = 0.f;
    if (p < P_) {
      int m = p % O_, ko = p / O_;                 // pair p = ko*36 + m
      float u = Ul[m * 512 + g];
      float v = UV[(size_t)(b * O_ + ko) * 1024 + 512 + g];
      hv = fmaxf(u + v + b1[g], 0.f);
    }
    hf[frag_idx(r, g, 512)] = f2bf(hv);
  }
  __syncthreads();

  const int w = tid >> 5, lane = tid & 31;
  v8f z = {0.f, 0.f, 0.f, 0.f, 0.f, 0.f, 0.f, 0.f};
  v8f acc[4][4];   // [mtile][ntile]; wave owns cols [w*64, w*64+64)
#pragma unroll
  for (int mi = 0; mi < 4; ++mi)
#pragma unroll
    for (int ni = 0; ni < 4; ++ni) acc[mi][ni] = z;

  auto LOAD = [&](v16bf (&a)[4], v16bf (&bv)[4], int kk) {
#pragma unroll
    for (int mi = 0; mi < 4; ++mi)
      a[mi] = *reinterpret_cast<const v16bf*>(hf + (((mi * 16 + kk) * 32 + lane) << 4));
#pragma unroll
    for (int ni = 0; ni < 4; ++ni)
      bv[ni] = *reinterpret_cast<const v16bf*>(
          W2f + ((((size_t)(w * 4 + ni) * 16 + kk) * 32 + lane) << 4));
  };
  auto MMA = [&](v16bf (&a)[4], v16bf (&bv)[4]) {
#pragma unroll
    for (int ni = 0; ni < 4; ++ni)
#pragma unroll
      for (int mi = 0; mi < 4; ++mi)
        acc[mi][ni] = wmma_bf16(a[mi], bv[ni], acc[mi][ni]);
  };

  v16bf a0[4], a1[4], b0[4], b1v[4];
  LOAD(a0, b0, 0);
  for (int kk = 0; kk < 14; kk += 2) {
    LOAD(a1, b1v, kk + 1);
    MMA(a0, b0);
    LOAD(a0, b0, kk + 2);
    MMA(a1, b1v);
  }
  LOAD(a1, b1v, 15);
  MMA(a0, b0);
  MMA(a1, b1v);

  // Epilogue: bias + ReLU, store x_pairs.
  const int nbase = w * 64 + (lane & 15);
  const int rsel  = (lane >> 4) * 8;
#pragma unroll
  for (int ni = 0; ni < 4; ++ni) {
    const int n = nbase + ni * 16;
    const float bias = b2[n];
#pragma unroll
    for (int mi = 0; mi < 4; ++mi)
#pragma unroll
      for (int i = 0; i < 8; ++i) {
        int p = p0 + mi * 16 + rsel + i;
        if (p < P_) {
          float x = fmaxf(acc[mi][ni][i] + bias, 0.f);
          xp[((size_t)b * P_ + p) * 512 + n] = x;
        }
      }
  }
}

// ---------- s = sum over pairs (deterministic) ----------

__global__ void k_reduce_s(const float* __restrict__ xp, float* __restrict__ s) {
  int idx = blockIdx.x * blockDim.x + threadIdx.x;  // 64*512
  if (idx >= B_ * 512) return;
  int b = idx >> 9, n = idx & 511;
  const float* base = xp + (size_t)b * P_ * 512 + n;
  float acc = 0.f;
  for (int p = 0; p < P_; ++p) acc += base[(size_t)p * 512];
  s[idx] = acc;
}

// ---------- BatchNorm over batch dim (training stats, biased var) ----------

__global__ void k_bn(const float* __restrict__ s, const float* __restrict__ gamma,
                     const float* __restrict__ beta, float* __restrict__ sn) {
  int c = threadIdx.x;   // 512 columns
  float m = 0.f;
  for (int b = 0; b < B_; ++b) m += s[b * 512 + c];
  m *= (1.f / B_);
  float v = 0.f;
  for (int b = 0; b < B_; ++b) { float d = s[b * 512 + c] - m; v += d * d; }
  v *= (1.f / B_);
  float rstd = rsqrtf(v + BN_EPS);
  float g = gamma[c], be = beta[c];
  for (int b = 0; b < B_; ++b)
    sn[b * 512 + c] = (s[b * 512 + c] - m) * rstd * g + be;
}

// ---------- small dense layers: out = relu(in @ W^T + bias), 64x512, K=512 ----------

__global__ void k_dense(const float* __restrict__ in, const float* __restrict__ W,
                        const float* __restrict__ bias, float* __restrict__ out) {
  int idx = blockIdx.x * blockDim.x + threadIdx.x;  // 64*512
  if (idx >= B_ * 512) return;
  int b = idx >> 9, j = idx & 511;
  const float* x = in + (size_t)b * 512;
  const float* wr = W + (size_t)j * 512;
  float acc = 0.f;
  for (int k = 0; k < 512; ++k) acc = fmaf(x[k], wr[k], acc);
  out[idx] = fmaxf(acc + bias[j], 0.f);
}

// ---------- launch ----------

extern "C" void kernel_launch(void* const* d_in, const int* in_sizes, int n_in,
                              void* d_out, int out_size, void* d_ws, size_t ws_size,
                              hipStream_t stream) {
  const float* objs  = (const float*)d_in[0];
  const float* W1    = (const float*)d_in[1];
  const float* b1    = (const float*)d_in[2];
  const float* W2    = (const float*)d_in[3];
  const float* b2    = (const float*)d_in[4];
  const float* gamma = (const float*)d_in[5];
  const float* beta  = (const float*)d_in[6];
  const float* W3    = (const float*)d_in[7];
  const float* b3    = (const float*)d_in[8];
  const float* W4    = (const float*)d_in[9];
  const float* b4    = (const float*)d_in[10];
  (void)in_sizes; (void)n_in; (void)out_size; (void)ws_size;

  char* ws = (char*)d_ws;
  u16*   objs_f = (u16*)(ws + 0);          // 2304*512  bf16 = 2359296 B
  u16*   w1_f   = (u16*)(ws + 2359296);    // 1024*512  bf16 = 1048576 B
  u16*   w2_f   = (u16*)(ws + 3407872);    //  512*512  bf16 =  524288 B
  float* UV     = (float*)(ws + 3932160);  // 2304*1024 f32  = 9437184 B
  float* s      = (float*)(ws + 13369344); // 64*512 f32
  float* sn     = (float*)(ws + 13500416); // 64*512 f32
  float* a1     = (float*)(ws + 13631488); // 64*512 f32

  float* out_agg = (float*)d_out;            // 64*512
  float* out_xp  = (float*)d_out + B_ * 512; // 64*1296*512

  k_pack_objs<<<4608, 256, 0, stream>>>(objs, objs_f);
  k_pack_w1  <<<2048, 256, 0, stream>>>(W1, w1_f);
  k_pack_w2  <<<1024, 256, 0, stream>>>(W2, w2_f);

  k_gemm1<<<dim3(36, 8), 256, 0, stream>>>(objs_f, w1_f, UV);

  k_gemm2<<<dim3(21, 64), 256, 65536 + 73728, stream>>>(UV, b1, w2_f, b2, out_xp);

  k_reduce_s<<<128, 256, 0, stream>>>(out_xp, s);
  k_bn<<<1, 512, 0, stream>>>(s, gamma, beta, sn);
  k_dense<<<128, 256, 0, stream>>>(sn, W3, b3, a1);
  k_dense<<<128, 256, 0, stream>>>(a1, W4, b4, out_agg);
}
